// LSoftmax_12893491823013
// MI455X (gfx1250) — matrix operand order repaired and checked by reading, not compiled, not run on hardware
//
#include <hip/hip_runtime.h>

typedef __attribute__((ext_vector_type(16))) _Float16 v16h;
typedef __attribute__((ext_vector_type(8)))  _Float16 v8h;
typedef __attribute__((ext_vector_type(8)))  float    v8f;

#define N_ROWS 8192
#define D_DIM  1024
#define C_CLS  1000
#define C_PAD  1024
#define INV_SQRT_T 1.8257418583505538f   /* 1/sqrt(0.3) */
#define LAM2_C 0.1f

// ---------- WMMA tile helpers (16x16x32 f16 -> f32) ----------
// A tile (16 rows x 32 K): lane m (0..15) holds row m,
// K = [k0..k0+15]; lane m+16 holds row m, K = [k0+16..k0+31].
__device__ __forceinline__ v16h ldA(const _Float16* __restrict__ F, int rbase, int k0, int lane) {
  int m = lane & 15, h = lane >> 4;
  return *(const v16h*)(F + (size_t)(rbase + m) * D_DIM + k0 + h * 16);
}
// B tile (32 K x 16 cols): lane n (0..15) holds column n, K=[k0..k0+15];
// lane n+16 holds column n, K=[k0+16..k0+31]. Columns are rows of the source matrix.
__device__ __forceinline__ v16h ldB(const _Float16* __restrict__ F, int cbase, int k0, int lane) {
  int n = lane & 15, h = lane >> 4;
  return *(const v16h*)(F + (size_t)(cbase + n) * D_DIM + k0 + h * 16);
}
__device__ __forceinline__ v8f wmma16(v16h a, v16h b, v8f c) {
  return __builtin_amdgcn_wmma_f32_16x16x32_f16(false, a, false, b, (short)0, c, false, false);
}

// ---------- init accumulators ----------
__global__ void k_init(float* acc) {
  if (threadIdx.x < 3) acc[threadIdx.x] = 0.0f;
}

// ---------- feature prep: norm + split-f16 (raw and normalized/scaled) ----------
__global__ void k_prep_feat(const float* __restrict__ feat,
                            _Float16* __restrict__ fhi, _Float16* __restrict__ flo,
                            _Float16* __restrict__ nhi, _Float16* __restrict__ nlo) {
  int row = blockIdx.x, t = threadIdx.x;
  const float* rp = feat + (size_t)row * D_DIM;
  float x[4]; float ss = 0.0f;
  #pragma unroll
  for (int i = 0; i < 4; i++) { x[i] = rp[t + 256 * i]; ss += x[i] * x[i]; }
  __shared__ float red[256];
  red[t] = ss; __syncthreads();
  for (int o = 128; o > 0; o >>= 1) { if (t < o) red[t] += red[t + o]; __syncthreads(); }
  float scale = rsqrtf(red[0]) * INV_SQRT_T;
  size_t base = (size_t)row * D_DIM;
  #pragma unroll
  for (int i = 0; i < 4; i++) {
    int j = t + 256 * i;
    float v = x[i];
    _Float16 h = (_Float16)v;
    fhi[base + j] = h;
    flo[base + j] = (_Float16)(v - (float)h);
    float vn = v * scale;
    _Float16 hn = (_Float16)vn;
    nhi[base + j] = hn;
    nlo[base + j] = (_Float16)(vn - (float)hn);
  }
}

// ---------- W prep: split-f16, pad rows [1000,1024) with zeros ----------
__global__ void k_prep_w(const float* __restrict__ W,
                         _Float16* __restrict__ whi, _Float16* __restrict__ wlo) {
  int c = blockIdx.x, t = threadIdx.x;
  size_t base = (size_t)c * D_DIM;
  #pragma unroll
  for (int i = 0; i < 4; i++) {
    int j = t + 256 * i;
    float v = (c < C_CLS) ? W[base + j] : 0.0f;
    _Float16 h = (_Float16)v;
    whi[base + j] = h;
    wlo[base + j] = (_Float16)(v - (float)h);
  }
}

// ---------- logits = feat @ W^T + b : one wave per 32x32 block (2x2 tiles) ----------
__global__ void k_logits(const _Float16* __restrict__ fhi, const _Float16* __restrict__ flo,
                         const _Float16* __restrict__ whi, const _Float16* __restrict__ wlo,
                         const float* __restrict__ b, float* __restrict__ out) {
  int lane  = threadIdx.x;          // blockDim.x == 32 (one wave)
  int cbase = blockIdx.x * 32;
  int rbase = blockIdx.y * 32;

  v8f acc[2][2];
  #pragma unroll
  for (int R = 0; R < 2; R++)
    #pragma unroll
    for (int C = 0; C < 2; C++) acc[R][C] = (v8f){};

  for (int k0 = 0; k0 < D_DIM; k0 += 32) {
    v16h ah[2], al[2], bh[2], bl[2];
    #pragma unroll
    for (int R = 0; R < 2; R++) {
      ah[R] = ldA(fhi, rbase + 16 * R, k0, lane);
      al[R] = ldA(flo, rbase + 16 * R, k0, lane);
    }
    #pragma unroll
    for (int C = 0; C < 2; C++) {
      bh[C] = ldB(whi, cbase + 16 * C, k0, lane);
      bl[C] = ldB(wlo, cbase + 16 * C, k0, lane);
    }
    #pragma unroll
    for (int R = 0; R < 2; R++)
      #pragma unroll
      for (int C = 0; C < 2; C++) {
        acc[R][C] = wmma16(ah[R], bh[C], acc[R][C]);
        acc[R][C] = wmma16(ah[R], bl[C], acc[R][C]);
        acc[R][C] = wmma16(al[R], bh[C], acc[R][C]);
      }
  }

  int nidx = lane & 15, half = lane >> 4;
  #pragma unroll
  for (int C = 0; C < 2; C++) {
    int col = cbase + 16 * C + nidx;
    if (col < C_CLS) {
      float bias = b[col];
      #pragma unroll
      for (int R = 0; R < 2; R++)
        #pragma unroll
        for (int v = 0; v < 8; v++) {
          int row = rbase + 16 * R + v + half * 8;
          out[(size_t)row * C_CLS + col] = acc[R][C][v] + bias;
        }
    }
  }
}

// ---------- cross entropy over logits rows ----------
__global__ void k_ce(const float* __restrict__ logits, const int* __restrict__ labels,
                     float* acc) {
  int row = blockIdx.x, t = threadIdx.x;
  const float* rp = logits + (size_t)row * C_CLS;
  __shared__ float red[256];
  float mx = -1e30f;
  for (int c = t; c < C_CLS; c += 256) mx = fmaxf(mx, rp[c]);
  red[t] = mx; __syncthreads();
  for (int o = 128; o > 0; o >>= 1) { if (t < o) red[t] = fmaxf(red[t], red[t + o]); __syncthreads(); }
  float rowmax = red[0]; __syncthreads();
  float s = 0.0f;
  for (int c = t; c < C_CLS; c += 256) s += __expf(rp[c] - rowmax);
  red[t] = s; __syncthreads();
  for (int o = 128; o > 0; o >>= 1) { if (t < o) red[t] += red[t + o]; __syncthreads(); }
  if (t == 0) {
    int lab = labels[row];
    float logp = rp[lab] - rowmax - __logf(red[0]);
    atomicAdd(&acc[0], -logp);
  }
}

// ---------- fused SCL: 32x64 register block per wave + online LSE ----------
__global__ void k_scl(const _Float16* __restrict__ nhi, const _Float16* __restrict__ nlo,
                      const int* __restrict__ labels, float* acc) {
  int t = threadIdx.x;
  int lane = t & 31, wave = t >> 5;     // 8 waves / block
  int rbase = blockIdx.x * 32;          // 32 rows per block
  int half = lane >> 4, nidx = lane & 15;

  int lab_r[2][8];
  #pragma unroll
  for (int R = 0; R < 2; R++)
    #pragma unroll
    for (int v = 0; v < 8; v++) lab_r[R][v] = labels[rbase + 16 * R + v + half * 8];

  float m[2][8], s[2][8], p[2][8], cn[2][8];
  #pragma unroll
  for (int R = 0; R < 2; R++)
    #pragma unroll
    for (int v = 0; v < 8; v++) { m[R][v] = -1e30f; s[R][v] = 0.0f; p[R][v] = 0.0f; cn[R][v] = 0.0f; }

  // 8 waves x 64 cols = 512 cols per superblock pass; 16 passes cover 8192 cols
  for (int sb = 0; sb < N_ROWS / 512; sb++) {
    int cbase0 = sb * 512 + wave * 64;

    v8f accm[2][4];
    #pragma unroll
    for (int R = 0; R < 2; R++)
      #pragma unroll
      for (int C = 0; C < 4; C++) accm[R][C] = (v8f){};

    for (int k0 = 0; k0 < D_DIM; k0 += 32) {
      v16h ah[2], al[2], bh[4], bl[4];
      #pragma unroll
      for (int R = 0; R < 2; R++) {
        ah[R] = ldA(nhi, rbase + 16 * R, k0, lane);
        al[R] = ldA(nlo, rbase + 16 * R, k0, lane);
      }
      #pragma unroll
      for (int C = 0; C < 4; C++) {
        bh[C] = ldB(nhi, cbase0 + 16 * C, k0, lane);
        bl[C] = ldB(nlo, cbase0 + 16 * C, k0, lane);
      }
      #pragma unroll
      for (int R = 0; R < 2; R++)
        #pragma unroll
        for (int C = 0; C < 4; C++) {
          accm[R][C] = wmma16(ah[R], bh[C], accm[R][C]);
          accm[R][C] = wmma16(ah[R], bl[C], accm[R][C]);
          accm[R][C] = wmma16(al[R], bh[C], accm[R][C]);
        }
    }

    // epilogue: online logsumexp + positive sums per row
    #pragma unroll
    for (int C = 0; C < 4; C++) {
      int gcol = cbase0 + 16 * C + nidx;
      int lab_c = labels[gcol];
      #pragma unroll
      for (int R = 0; R < 2; R++)
        #pragma unroll
        for (int v = 0; v < 8; v++) {
          int grow = rbase + 16 * R + v + half * 8;
          if (gcol != grow) {                  // diagonal excluded (-inf in reference)
            float x = accm[R][C][v];
            float mn = fmaxf(m[R][v], x);
            s[R][v] = s[R][v] * __expf(m[R][v] - mn) + __expf(x - mn);
            m[R][v] = mn;
            if (lab_c == lab_r[R][v]) { p[R][v] += x; cn[R][v] += 1.0f; }
          }
        }
    }
  }

  // reduce across the 16 lanes of each half-wave (columns of the tile)
  #pragma unroll
  for (int mask = 1; mask < 16; mask <<= 1) {
    #pragma unroll
    for (int R = 0; R < 2; R++)
      #pragma unroll
      for (int v = 0; v < 8; v++) {
        float m2 = __shfl_xor(m[R][v], mask, 32);
        float s2 = __shfl_xor(s[R][v], mask, 32);
        float p2 = __shfl_xor(p[R][v], mask, 32);
        float c2 = __shfl_xor(cn[R][v], mask, 32);
        float mn = fmaxf(m[R][v], m2);
        s[R][v] = s[R][v] * __expf(m[R][v] - mn) + s2 * __expf(m2 - mn);
        m[R][v] = mn;
        p[R][v] += p2; cn[R][v] += c2;
      }
  }

  __shared__ float sm[8][32], ssum[8][32], sp[8][32], sc[8][32];
  if (nidx == 0) {
    #pragma unroll
    for (int R = 0; R < 2; R++)
      #pragma unroll
      for (int v = 0; v < 8; v++) {
        int r = 16 * R + v + half * 8;
        sm[wave][r] = m[R][v]; ssum[wave][r] = s[R][v];
        sp[wave][r] = p[R][v]; sc[wave][r] = cn[R][v];
      }
  }
  __syncthreads();

  if (t < 32) {   // one thread per row of this 32-row block: merge 8 wave partials
    float M = -1e30f, S = 0.0f, P = 0.0f, CN = 0.0f;
    #pragma unroll
    for (int w = 0; w < 8; w++) {
      float m2 = sm[w][t], s2 = ssum[w][t];
      float mn = fmaxf(M, m2);
      S = S * __expf(M - mn) + s2 * __expf(m2 - mn);
      M = mn;
      P += sp[w][t]; CN += sc[w][t];
    }
    float bottom = M + __logf(S);          // logsumexp over j != i of sim/t
    if (CN > 0.5f) {
      float per = -(P - CN * bottom) / CN; // -mean over positives of (sim - lse)
      atomicAdd(&acc[1], per);
      atomicAdd(&acc[2], 1.0f);
    }
  }
}

// ---------- combine losses ----------
__global__ void k_final(const float* acc, float* out) {
  float ce  = acc[0] / (float)N_ROWS;
  float scl = (acc[2] > 0.5f) ? acc[1] / acc[2] : 0.0f;
  out[(size_t)N_ROWS * C_CLS] = (1.0f - LAM2_C) * ce + LAM2_C * scl;
}

extern "C" void kernel_launch(void* const* d_in, const int* in_sizes, int n_in,
                              void* d_out, int out_size, void* d_ws, size_t ws_size,
                              hipStream_t stream) {
  const float* feat   = (const float*)d_in[0];
  const int*   labels = (const int*)d_in[1];
  const float* W      = (const float*)d_in[2];
  const float* b      = (const float*)d_in[3];
  float* out = (float*)d_out;

  size_t ND = (size_t)N_ROWS * D_DIM;
  size_t WD = (size_t)C_PAD * D_DIM;
  _Float16* nhi = (_Float16*)d_ws;   // normalized*1/sqrt(t), hi
  _Float16* nlo = nhi + ND;          // lo residual
  _Float16* fhi = nlo + ND;          // raw feat hi
  _Float16* flo = fhi + ND;          // raw feat lo
  _Float16* whi = flo + ND;          // W hi (padded to 1024 rows)
  _Float16* wlo = whi + WD;          // W lo
  float* acc = (float*)(wlo + WD);   // [0]=ce_sum [1]=scl_sum [2]=scl_valid

  k_init<<<1, 32, 0, stream>>>(acc);
  k_prep_feat<<<N_ROWS, 256, 0, stream>>>(feat, fhi, flo, nhi, nlo);
  k_prep_w<<<C_PAD, 256, 0, stream>>>(W, whi, wlo);
  k_logits<<<dim3(C_PAD / 32, N_ROWS / 32), 32, 0, stream>>>(fhi, flo, whi, wlo, b, out);
  k_ce<<<N_ROWS, 256, 0, stream>>>(out, labels, acc);
  k_scl<<<N_ROWS / 32, 256, 0, stream>>>(nhi, nlo, labels, acc);
  k_final<<<1, 1, 0, stream>>>(acc, out);
}
